// VectorQuantizer_46823733461505
// MI455X (gfx1250) — matrix operand order repaired and checked by reading, not compiled, not run on hardware
//
#include <hip/hip_runtime.h>
#include <hip/hip_bf16.h>
#include <math.h>

typedef __attribute__((ext_vector_type(16))) __bf16 v16bf;
typedef __attribute__((ext_vector_type(8)))  float  v8f;

#define NUM_EMB   2048
#define EMB_DIM   64
#define N_TOKENS  65536            // 16*64*64
#define N_ELEMS   4194304          // N_TOKENS * EMB_DIM
#define ROWS_PER_WAVE  32
#define ROWS_PER_BLOCK 128         // 4 waves * 32 rows
#define N_TILES   (NUM_EMB / 16)   // 128

// workspace layout (bytes)
#define WS_SQSUM_OFF   0
#define WS_COUNTS_OFF  1024
#define WS_CNORM_OFF   (1024 + NUM_EMB * 4)            // 9216
#define WS_PACKED_OFF  (WS_CNORM_OFF + NUM_EMB * 4)    // 17408 (128B aligned)

__device__ __forceinline__ __bf16 f2bf(float f) {
    unsigned u = __builtin_bit_cast(unsigned, f);
    unsigned r = (u + 0x7FFFu + ((u >> 16) & 1u)) >> 16;   // round-to-nearest-even
    return __builtin_bit_cast(__bf16, (unsigned short)r);
}

// Build the two A fragments (K 0..31 and K 32..63) for one 16-row group.
// A striping (this code's contiguous k order, 16-bit A 16x32): row M = lane%16; koff = 16*(lane>=16);
// a0 elems 0-15: K=koff+j ; a1: +32.
__device__ __forceinline__ void build_a_frags(const float* __restrict__ xr,
                                              int koff, v16bf& a0, v16bf& a1) {
    const float4* c0 = (const float4*)(xr + koff);
    const float4* c1 = (const float4*)(xr + 8 + koff);
    const float4* c2 = (const float4*)(xr + 32 + koff);
    const float4* c3 = (const float4*)(xr + 40 + koff);
    float4 f0a = c0[0], f0b = c0[1], f1a = c1[0], f1b = c1[1];
    float4 f2a = c2[0], f2b = c2[1], f3a = c3[0], f3b = c3[1];
    a0[0]=f2bf(f0a.x); a0[1]=f2bf(f0a.y); a0[2]=f2bf(f0a.z); a0[3]=f2bf(f0a.w);
    a0[4]=f2bf(f0b.x); a0[5]=f2bf(f0b.y); a0[6]=f2bf(f0b.z); a0[7]=f2bf(f0b.w);
    a0[8]=f2bf(f1a.x); a0[9]=f2bf(f1a.y); a0[10]=f2bf(f1a.z); a0[11]=f2bf(f1a.w);
    a0[12]=f2bf(f1b.x); a0[13]=f2bf(f1b.y); a0[14]=f2bf(f1b.z); a0[15]=f2bf(f1b.w);
    a1[0]=f2bf(f2a.x); a1[1]=f2bf(f2a.y); a1[2]=f2bf(f2a.z); a1[3]=f2bf(f2a.w);
    a1[4]=f2bf(f2b.x); a1[5]=f2bf(f2b.y); a1[6]=f2bf(f2b.z); a1[7]=f2bf(f2b.w);
    a1[8]=f2bf(f3a.x); a1[9]=f2bf(f3a.y); a1[10]=f2bf(f3a.z); a1[11]=f2bf(f3a.w);
    a1[12]=f2bf(f3b.x); a1[13]=f2bf(f3b.y); a1[14]=f2bf(f3b.z); a1[15]=f2bf(f3b.w);
}

// ---------------------------------------------------------------------------
// Pre-pass: zero counts/sqsum, compute ||c||^2, pack -2*c (bf16) into the
// exact WMMA B-fragment lane layout: per 16-code tile t, lane l holds 32
// contiguous bf16: cols n = t*16 + (l&15); lanes 0-15 cover K 0-15 (b0) /
// 32-47 (b1), lanes 16-31 cover K 16-31 / 48-63  (contiguous k order, same as A).
// ---------------------------------------------------------------------------
__global__ void vq_prep_kernel(const float* __restrict__ emb,
                               unsigned char* __restrict__ ws) {
    int n = blockIdx.x * blockDim.x + threadIdx.x;   // code id, 0..2047
    if (n >= NUM_EMB) return;
    float*          sqsum  = (float*)(ws + WS_SQSUM_OFF);
    unsigned*       counts = (unsigned*)(ws + WS_COUNTS_OFF);
    float*          cnorm  = (float*)(ws + WS_CNORM_OFF);
    unsigned short* packed = (unsigned short*)(ws + WS_PACKED_OFF);

    if (n == 0) *sqsum = 0.0f;
    counts[n] = 0u;

    const float* e = emb + (size_t)n * EMB_DIM;
    float nrm = 0.0f;
    #pragma unroll
    for (int k = 0; k < EMB_DIM; ++k) nrm += e[k] * e[k];
    cnorm[n] = nrm;

    int t   = n >> 4;          // tile
    int col = n & 15;          // column within tile (== lane % 16)
    unsigned short* lo = packed + (size_t)t * 1024 + (size_t)col * 32;        // lane = col
    unsigned short* hi = packed + (size_t)t * 1024 + (size_t)(col + 16) * 32; // lane = col+16
    #pragma unroll
    for (int j = 0; j < 16; ++j) {
        lo[j]      = __builtin_bit_cast(unsigned short, f2bf(-2.0f * e[j]));        // K = j
        lo[j + 16] = __builtin_bit_cast(unsigned short, f2bf(-2.0f * e[32 + j]));   // K = 32+j
        hi[j]      = __builtin_bit_cast(unsigned short, f2bf(-2.0f * e[16 + j]));   // K = 16+j
        hi[j + 16] = __builtin_bit_cast(unsigned short, f2bf(-2.0f * e[48 + j]));   // K = 48+j
    }
}

// ---------------------------------------------------------------------------
// Main fused kernel: per block 4 waves x 32 rows; each wave runs two
// independent WMMA accumulator chains (no D->C RAW stalls) over all 2048
// codes with in-register running argmin, then gather/quantize/loss epilogue.
// ---------------------------------------------------------------------------
__global__ void __launch_bounds__(128)
vq_main_kernel(const float* __restrict__ x,
               const float* __restrict__ emb,
               unsigned char* __restrict__ ws,
               float* __restrict__ q_out,
               float* __restrict__ idx_out) {
    const float*          cnorm  = (const float*)(ws + WS_CNORM_OFF);
    const unsigned short* packed = (const unsigned short*)(ws + WS_PACKED_OFF);
    unsigned*             counts = (unsigned*)(ws + WS_COUNTS_OFF);
    float*                sqsum  = (float*)(ws + WS_SQSUM_OFF);

    const int lane  = threadIdx.x & 31;
    const int wave  = threadIdx.x >> 5;
    const int half  = lane >> 4;          // 0 or 1
    const int cm    = lane & 15;
    const int rbase = blockIdx.x * ROWS_PER_BLOCK + wave * ROWS_PER_WAVE;

    // ---- build A fragments for two 16-row groups ----
    const int koff = half * 16;
    v16bf a0, a1, a2, a3;
    build_a_frags(x + (size_t)(rbase + cm) * EMB_DIM,      koff, a0, a1);
    build_a_frags(x + (size_t)(rbase + 16 + cm) * EMB_DIM, koff, a2, a3);

    // ---- GEMM over all codes, fused running argmin (16 rows tracked/lane) ----
    float minv[16];
    int   mini[16];
    #pragma unroll
    for (int r = 0; r < 16; ++r) { minv[r] = __builtin_inff(); mini[r] = 0; }

    const unsigned short* pl = packed + (size_t)lane * 32;
    #pragma unroll 2
    for (int t = 0; t < N_TILES; ++t) {
        const v16bf* bp = (const v16bf*)(pl + (size_t)t * 1024);
        v16bf b0 = bp[0];
        v16bf b1 = bp[1];
        float cn = cnorm[t * 16 + cm];
        v8f z = {0.f,0.f,0.f,0.f,0.f,0.f,0.f,0.f};
        v8f acc0 = __builtin_amdgcn_wmma_f32_16x16x32_bf16(false, a0, false, b0,
                                                           (short)0, z, false, false);
        v8f acc1 = __builtin_amdgcn_wmma_f32_16x16x32_bf16(false, a2, false, b0,
                                                           (short)0, z, false, false);
        acc0 = __builtin_amdgcn_wmma_f32_16x16x32_bf16(false, a1, false, b1,
                                                       (short)0, acc0, false, false);
        acc1 = __builtin_amdgcn_wmma_f32_16x16x32_bf16(false, a3, false, b1,
                                                       (short)0, acc1, false, false);
        const int n = t * 16 + cm;
        #pragma unroll
        for (int r = 0; r < 8; ++r) {
            float s0 = acc0[r] + cn;        // ||c||^2 - 2 x.c   (rows 0..15)
            if (s0 < minv[r])     { minv[r] = s0;     mini[r] = n; }
            float s1 = acc1[r] + cn;        //                    (rows 16..31)
            if (s1 < minv[8 + r]) { minv[8 + r] = s1; mini[8 + r] = n; }
        }
    }

    // ---- cross-lane argmin over the 16 columns (stays within half-wave) ----
    #pragma unroll
    for (int r = 0; r < 16; ++r) {
        float v = minv[r]; int i = mini[r];
        #pragma unroll
        for (int m = 1; m < 16; m <<= 1) {
            float ov = __shfl_xor(v, m);
            int   oi = __shfl_xor(i, m);
            if (ov < v || (ov == v && oi < i)) { v = ov; i = oi; }
        }
        minv[r] = v; mini[r] = i;
    }
    // lanes cm==0 (lane 0 / lane 16): reg r<8 -> group0 row 8*half+r,
    //                                 reg 8+r -> group1 row 16+8*half+r

    // ---- epilogue: gather quantized rows, write idx, histogram, loss ----
    float errsum = 0.0f;
    #pragma unroll 4
    for (int m = 0; m < ROWS_PER_WAVE; ++m) {
        const int g   = m >> 4;                  // row group
        const int mr  = m & 15;                  // row within group
        const int src = (mr >> 3) * 16;          // lane holding this row
        const int r   = g * 8 + (mr & 7);        // register index
        const int idx = __shfl(mini[r], src);
        const int grow = rbase + m;
        const float2* ef = (const float2*)(emb + (size_t)idx * EMB_DIM);
        const float2* xf = (const float2*)(x + (size_t)grow * EMB_DIM);
        float2 qv = ef[lane];
        float2 xv = xf[lane];
        ((float2*)(q_out + (size_t)grow * EMB_DIM))[lane] = qv;
        float d0 = qv.x - xv.x, d1 = qv.y - xv.y;
        errsum += d0 * d0 + d1 * d1;
        if (lane == 0) {
            idx_out[grow] = (float)idx;
            atomicAdd(&counts[idx], 1u);
        }
    }
    // wave-reduce squared error, one atomic per wave
    #pragma unroll
    for (int m = 16; m >= 1; m >>= 1) errsum += __shfl_xor(errsum, m);
    if (lane == 0) atomicAdd(sqsum, errsum);
}

// ---------------------------------------------------------------------------
// Finalize: avg_probs, entropy, perplexity, scalar losses.
// ---------------------------------------------------------------------------
__global__ void vq_final_kernel(const unsigned char* __restrict__ ws,
                                float* __restrict__ out) {
    const unsigned* counts = (const unsigned*)(ws + WS_COUNTS_OFF);
    const float*    sqsum  = (const float*)(ws + WS_SQSUM_OFF);
    float* vq_loss  = out + N_ELEMS;
    float* commit   = vq_loss + 1;
    float* codebook = commit + 1;
    float* perp     = codebook + 1;
    float* entl     = perp + 1;
    float* avgp     = entl + 1 + N_TOKENS;

    __shared__ float red[256];
    const int tid = threadIdx.x;
    float ep = 0.0f;
    for (int k = tid; k < NUM_EMB; k += 256) {
        float p = (float)counts[k] * (1.0f / (float)N_TOKENS);
        avgp[k] = p;
        ep += p * logf(p + 1e-5f);
    }
    red[tid] = ep;
    __syncthreads();
    for (int s = 128; s > 0; s >>= 1) {
        if (tid < s) red[tid] += red[tid + s];
        __syncthreads();
    }
    if (tid == 0) {
        float entropy = red[0];                         // sum p log(p+eps)
        float L = sqsum[0] * (1.0f / (float)N_ELEMS);   // mean (q-x)^2
        *commit   = L;
        *codebook = L;
        *entl     = entropy;
        *perp     = expf(-entropy);
        *vq_loss  = L + 0.25f * L + 0.1f * entropy;
    }
}

extern "C" void kernel_launch(void* const* d_in, const int* in_sizes, int n_in,
                              void* d_out, int out_size, void* d_ws, size_t ws_size,
                              hipStream_t stream) {
    const float* x   = (const float*)d_in[0];
    const float* emb = (const float*)d_in[1];
    float* out = (float*)d_out;
    unsigned char* ws = (unsigned char*)d_ws;

    float* q_out   = out;                              // [16,64,64,64]
    float* idx_out = out + N_ELEMS + 5;                // after 5 scalars

    vq_prep_kernel<<<(NUM_EMB + 255) / 256, 256, 0, stream>>>(emb, ws);
    vq_main_kernel<<<N_TOKENS / ROWS_PER_BLOCK, 128, 0, stream>>>(x, emb, ws, q_out, idx_out);
    vq_final_kernel<<<1, 256, 0, stream>>>(ws, out);
}
